// PNA_aff_24756191494679
// MI455X (gfx1250) — compile-verified, matrix operand order checked by reading, not run on hardware
//
#include <hip/hip_runtime.h>
#include <hip/hip_bf16.h>

#define NN   8000
#define NE   100000
#define NG   16
#define H    80
#define TW   5
#define FO   16
#define KP   96        // 80 padded to mult of 32
#define KPOST 1056     // 1040 padded to mult of 32
#define THW  (TW*H)    // 400

typedef __attribute__((ext_vector_type(16))) _Float16 v16h;
typedef __attribute__((ext_vector_type(8)))  _Float16 v8h;
typedef __attribute__((ext_vector_type(8)))  float    v8f;
typedef _Float16 h16;

// ---------- helpers ----------
__device__ __forceinline__ unsigned fenc(float f) {
    unsigned u = __float_as_uint(f);
    return (u >> 31) ? ~u : (u | 0x80000000u);
}
__device__ __forceinline__ float fdec(unsigned e) {
    unsigned u = (e >> 31) ? (e ^ 0x80000000u) : ~e;
    return __uint_as_float(u);
}

// ISA 7.12.2: 16-bit A 16x32 per-lane K map (used by the swizzle kernels)
__device__ __forceinline__ int kmap(int j, int half) {
    return (j < 8) ? (half * 8 + j) : (16 + half * 8 + (j - 8));
}

// A-fragment from a row-major f16 matrix: two contiguous 16B loads per lane.
// element j<8  -> K = k0 + 8*half + j
// element j>=8 -> K = k0 + 16 + 8*half + (j-8)
__device__ __forceinline__ v16h load_frag_rm(const h16* A, int lda, int r, int half, int k0) {
    const h16* base = A + (size_t)r * lda + k0 + half * 8;
    v8h lo = *(const v8h*)(base);
    v8h hi = *(const v8h*)(base + 16);
    return __builtin_shufflevector(lo, hi, 0, 1, 2, 3, 4, 5, 6, 7, 8, 9, 10, 11, 12, 13, 14, 15);
}
// B-fragment from a pre-swizzled buffer laid out [tile][lane][16]
__device__ __forceinline__ v16h load_frag_sw(const h16* Bs, int tile, int lane) {
    const h16* base = Bs + (size_t)tile * 512 + lane * 16;
    v8h lo = *(const v8h*)(base);
    v8h hi = *(const v8h*)(base + 8);
    return __builtin_shufflevector(lo, hi, 0, 1, 2, 3, 4, 5, 6, 7, 8, 9, 10, 11, 12, 13, 14, 15);
}

// ---------- utility kernels ----------
__global__ void k_fill_u32(unsigned* p, unsigned v, int n) {
    int i = blockIdx.x * blockDim.x + threadIdx.x;
    if (i < n) p[i] = v;
}

__global__ void k_node_pre(const float* x, const float* preW, const float* preb,
                           float* op, h16* oph) {
    int i = blockIdx.x * blockDim.x + threadIdx.x;
    if (i >= NN * KP) return;
    int n = i / KP, f = i % KP;
    if (f < H) {
        float v = x[n * 2] * preW[f] + preb[f];
        op[(size_t)n * H + f] = v;
        oph[(size_t)n * KP + f] = (h16)v;
    } else {
        oph[(size_t)n * KP + f] = (h16)0.f;
    }
}

__global__ void k_deg(const int* ei, float* deg) {
    int e = blockIdx.x * blockDim.x + threadIdx.x;
    if (e < NE) atomicAdd(&deg[ei[NE + e]], 1.0f);
}

__global__ void k_delta(const float* deg, float* delta) {
    __shared__ float red[256];
    float s = 0.f;
    for (int n = threadIdx.x; n < NN; n += 256) s += logf(deg[n] + 1.0f);
    red[threadIdx.x] = s;
    __syncthreads();
    for (int o = 128; o > 0; o >>= 1) {
        if (threadIdx.x < o) red[threadIdx.x] += red[threadIdx.x + o];
        __syncthreads();
    }
    if (threadIdx.x == 0) delta[0] = red[0] / (float)NN;
}

__global__ void k_scalars(const float* deg, const float* delta,
                          float* amp, float* att, float* hnb) {
    int n = blockIdx.x * blockDim.x + threadIdx.x;
    if (n >= NN) return;
    float d = deg[n];
    float dc = fmaxf(d, 1.0f);
    float L = logf(dc + 1.0f);
    float de = delta[0];
    amp[n] = L / de;
    att[n] = de / L;
    hnb[n] = (d > 0.f) ? 1.f : 0.f;
}

// ---------- weight conversion: f32 row-major -> f16 WMMA-fragment-swizzled ----------
// Layout: [..tiles..][lane(32)][j(16)], tile = linearized (.., coltile, kchunk)
__global__ void k_cvt_pre(const float* W, h16* wi, h16* wj) {  // W: [TW][2H][H]
    int i = blockIdx.x * blockDim.x + threadIdx.x;
    if (i >= TW * 5 * 3 * 512) return;
    int j = i & 15, lane = (i >> 4) & 31, kc = (i >> 9) % 3, ct = (i / 1536) % 5, t = i / 7680;
    int half = lane >> 4, r = lane & 15;
    int k = kc * 32 + kmap(j, half);
    int g = ct * 16 + r;
    wi[i] = (k < H) ? (h16)W[(size_t)t * 2 * H * H + (size_t)k * H + g] : (h16)0.f;
    wj[i] = (k < H) ? (h16)W[(size_t)t * 2 * H * H + (size_t)(H + k) * H + g] : (h16)0.f;
}
__global__ void k_cvt_post(const float* W, h16* wp) {          // W: [TW][1040][16]
    int i = blockIdx.x * blockDim.x + threadIdx.x;
    if (i >= TW * 33 * 512) return;
    int j = i & 15, lane = (i >> 4) & 31, kc = (i >> 9) % 33, t = i / 16896;
    int half = lane >> 4, r = lane & 15;
    int k = kc * 32 + kmap(j, half);
    wp[i] = (k < 13 * H) ? (h16)W[(size_t)t * 13 * H * FO + (size_t)k * FO + r] : (h16)0.f;
}
__global__ void k_cvt_lin(const float* W, h16* wl) {           // W: [80][80]
    int i = blockIdx.x * blockDim.x + threadIdx.x;
    if (i >= 5 * 3 * 512) return;
    int j = i & 15, lane = (i >> 4) & 31, kc = (i >> 9) % 3, ct = i / 1536;
    int half = lane >> 4, r = lane & 15;
    int k = kc * 32 + kmap(j, half);
    int g = ct * 16 + r;
    wl[i] = (k < H) ? (h16)W[(size_t)k * H + g] : (h16)0.f;
}

// ---------- GEMM: hi/hj = op @ Wi/Wj  (per tower), WMMA ----------
__global__ void k_gemm_hihj(const h16* __restrict__ oph, const h16* __restrict__ wih,
                            const h16* __restrict__ wjh, float* __restrict__ hi,
                            float* __restrict__ hj) {
    const int n0 = blockIdx.x * 16;
    const int ct = blockIdx.y;
    const int t  = blockIdx.z >> 1;
    const int wj = blockIdx.z & 1;
    const int lane = threadIdx.x;
    const int r = lane & 15, half = lane >> 4;
    const h16* A  = oph + (size_t)n0 * KP;
    const h16* Bs = (wj ? wjh : wih);
    const int tile0 = (t * 5 + ct) * 3;
    float* D = wj ? hj : hi;
    v8f acc = {};
#pragma unroll
    for (int kc = 0; kc < 3; ++kc) {
        v16h a = load_frag_rm(A, KP, r, half, kc * 32);
        v16h b = load_frag_sw(Bs, tile0 + kc, lane);
        acc = __builtin_amdgcn_wmma_f32_16x16x32_f16(false, a, false, b, (short)0, acc,
                                                     false, false);
    }
#pragma unroll
    for (int i = 0; i < 8; ++i)
        D[(size_t)(n0 + i + 8 * half) * THW + t * H + ct * 16 + r] = acc[i];
}

// ---------- edge scatter: aggregate hj[src] into per-target stats ----------
__global__ void k_edge_scatter(const int* __restrict__ ei, const float* __restrict__ hj,
                               float* __restrict__ sb, float* __restrict__ s2b,
                               unsigned* __restrict__ mnb, unsigned* __restrict__ mxb) {
    int idx = blockIdx.x * blockDim.x + threadIdx.x;
    if (idx >= NE * (THW / 4)) return;
    int e  = idx / (THW / 4);
    int fb = (idx % (THW / 4)) * 4;
    int s = ei[e], t = ei[NE + e];
    const float4 v4 = *(const float4*)(hj + (size_t)s * THW + fb);
    float v[4] = {v4.x, v4.y, v4.z, v4.w};
    size_t o = (size_t)t * THW + fb;
#pragma unroll
    for (int c = 0; c < 4; ++c) {
        atomicAdd(&sb[o + c], v[c]);
        atomicAdd(&s2b[o + c], v[c] * v[c]);
        atomicMin(&mnb[o + c], fenc(v[c]));
        atomicMax(&mxb[o + c], fenc(v[c]));
    }
}

// ---------- fused aggregate build + post GEMM (K=1040->1056), WMMA ----------
__global__ void k_post(const float* __restrict__ op, const float* __restrict__ hi,
                       const float* __restrict__ preb, const float* __restrict__ deg,
                       const float* __restrict__ amp, const float* __restrict__ att,
                       const float* __restrict__ hnb, const float* __restrict__ sb,
                       const float* __restrict__ s2b, const unsigned* __restrict__ mnb,
                       const unsigned* __restrict__ mxb, const h16* __restrict__ wpost,
                       const float* __restrict__ postb, h16* __restrict__ outpost) {
    __shared__ float sOp[16 * H];
    __shared__ float sAgg[16 * 4 * H];
    __shared__ float sAmp[16], sAtt[16], sDeg[16], sHnb[16];
    __shared__ h16   sA[16 * KPOST];
    const int n0 = blockIdx.x * 16;
    const int t  = blockIdx.y;
    const int tid = threadIdx.x;
    if (tid < 16) {
        int node = n0 + tid;
        sAmp[tid] = amp[node]; sAtt[tid] = att[node];
        sDeg[tid] = deg[node]; sHnb[tid] = hnb[node];
    }
    // Async-copy the contiguous 16x80 f32 op tile into LDS (CDNA5 async path,
    // tracked by ASYNCcnt): 10 issues x 32 lanes x 16B = 5120B.
    {
        unsigned lbase = (unsigned)(size_t)(void*)sOp;
        const float* gbase = op + (size_t)n0 * H;
#pragma unroll
        for (int it = 0; it < 10; ++it) {
            int elem = (it * 32 + tid) * 4;          // float index in tile
            unsigned laddr = lbase + elem * 4;       // LDS byte address
            const void* ga = (const void*)(gbase + elem);
            asm volatile("global_load_async_to_lds_b128 %0, %1, off"
                         :: "v"(laddr), "v"(ga) : "memory");
        }
        asm volatile("s_wait_asynccnt 0" ::: "memory");
    }
    __syncthreads();
    for (int e = tid; e < 16 * 4 * H; e += 32) {
        int m = e / (4 * H), idx = e % (4 * H), a = idx / H, g = idx % H;
        size_t o = (size_t)(n0 + m) * THW + t * H + g;
        float c  = hi[o] + preb[t * H + g];
        float S  = sb[o], S2 = s2b[o];
        float dm = sDeg[m], d = fmaxf(dm, 1.0f);
        float mean = (dm * c + S) / d;
        float v;
        if (a == 0)      v = mean;
        else if (a == 1) v = (sHnb[m] > 0.f) ? (c + fdec(mnb[o])) : 0.f;
        else if (a == 2) v = (sHnb[m] > 0.f) ? (c + fdec(mxb[o])) : 0.f;
        else {
            float msq = (dm * c * c + 2.f * c * S + S2) / d;
            v = sqrtf(fmaxf(msq - mean * mean, 0.f) + 1e-5f);
        }
        sAgg[e] = v;
    }
    __syncthreads();
    for (int e = tid; e < 16 * KPOST; e += 32) {
        int m = e / KPOST, k = e % KPOST;
        float v;
        if (k < H) v = sOp[m * H + k];
        else if (k < 13 * H) {
            int j = k - H, grp = j / (4 * H), idx = j % (4 * H);
            float sc = (grp == 0) ? 1.f : ((grp == 1) ? sAmp[m] : sAtt[m]);
            v = sAgg[m * 4 * H + idx] * sc;
        } else v = 0.f;
        sA[e] = (h16)v;
    }
    __syncthreads();
    const int r = tid & 15, half = tid >> 4;
    v8f acc = {};
#pragma unroll
    for (int kc = 0; kc < 33; ++kc) {
        __builtin_prefetch(wpost + ((size_t)t * 33 + kc + 1) * 512, 0, 1);
        v16h a = load_frag_rm(sA, KPOST, r, half, kc * 32);
        v16h b = load_frag_sw(wpost, t * 33 + kc, tid);
        acc = __builtin_amdgcn_wmma_f32_16x16x32_f16(false, a, false, b, (short)0, acc,
                                                     false, false);
    }
#pragma unroll
    for (int i = 0; i < 8; ++i) {
        int node = n0 + i + 8 * half;
        outpost[(size_t)node * KP + t * FO + r] = (h16)(acc[i] + postb[t * FO + r]);
    }
}

// ---------- conv_lin GEMM (80x80), WMMA ----------
__global__ void k_gemm_lin(const h16* __restrict__ A0, const h16* __restrict__ wl,
                           const float* __restrict__ lb, float* __restrict__ out2) {
    const int n0 = blockIdx.x * 16;
    const int ct = blockIdx.y;
    const int lane = threadIdx.x;
    const int r = lane & 15, half = lane >> 4;
    const h16* A = A0 + (size_t)n0 * KP;
    v8f acc = {};
#pragma unroll
    for (int kc = 0; kc < 3; ++kc) {
        v16h a = load_frag_rm(A, KP, r, half, kc * 32);
        v16h b = load_frag_sw(wl, ct * 3 + kc, lane);
        acc = __builtin_amdgcn_wmma_f32_16x16x32_f16(false, a, false, b, (short)0, acc,
                                                     false, false);
    }
#pragma unroll
    for (int i = 0; i < 8; ++i)
        out2[(size_t)(n0 + i + 8 * half) * H + ct * 16 + r] = acc[i] + lb[ct * 16 + r];
}

// ---------- batch norm ----------
__global__ void k_bnstat(const float* out2, float* st) {
    int i = blockIdx.x * blockDim.x + threadIdx.x;
    if (i >= NN * H) return;
    int f = i % H;
    float v = out2[i];
    atomicAdd(&st[f], v);
    atomicAdd(&st[H + f], v * v);
}
__global__ void k_bnorm(const float* out2, const float* st, const float* gamma,
                        const float* beta, float* op, h16* oph) {
    int i = blockIdx.x * blockDim.x + threadIdx.x;
    if (i >= NN * KP) return;
    int n = i / KP, f = i % KP;
    if (f < H) {
        float mu  = st[f] / (float)NN;
        float var = st[H + f] / (float)NN - mu * mu;
        float v = gamma[f] * (out2[(size_t)n * H + f] - mu) * rsqrtf(var + 1e-5f) + beta[f];
        v = fmaxf(v, 0.f);
        op[(size_t)n * H + f] = v;
        oph[(size_t)n * KP + f] = (h16)v;
    } else {
        oph[(size_t)n * KP + f] = (h16)0.f;
    }
}

// ---------- tail: xc, log-sum scatter, pooling, MLP ----------
__global__ void k_xc(const float* op, const float* x, float* xc) {
    int i = blockIdx.x * blockDim.x + threadIdx.x;
    if (i >= NN * 40) return;
    int n = i / 40, j = i % 40;
    xc[i] = op[(size_t)n * H + j] * x[n * 2 + 1] + op[(size_t)n * H + 40 + j];
}
__global__ void k_logsum(const int* ei, const float* xc, float* ls) {
    int i = blockIdx.x * blockDim.x + threadIdx.x;
    if (i >= NE * 40) return;
    int e = i / 40, j = i % 40;
    int s = ei[e], t = ei[NE + e];
    atomicAdd(&ls[(size_t)t * 40 + j], logf(xc[(size_t)s * 40 + j] + 1e-6f));
}
__global__ void k_pool(const float* xc, const float* ls, const int* batch,
                       float* gs, float* gc) {
    int n = blockIdx.x * blockDim.x + threadIdx.x;
    if (n >= NN) return;
    int b = batch[n];
    atomicAdd(&gc[b], 1.0f);
    for (int j = 0; j < 40; ++j) {
        float v = expf(ls[(size_t)n * 40 + j] + logf(xc[(size_t)n * 40 + j] + 1e-6f));
        atomicAdd(&gs[(size_t)b * 40 + j], v);
    }
}
__global__ void k_final(const float* gs, const float* gc, const float* mlpW,
                        const float* mlpb, float* out) {
    int g = threadIdx.x;
    if (g >= NG) return;
    float acc = mlpb[0];
    float inv = 1.0f / fmaxf(gc[g], 1.0f);
    for (int j = 0; j < 40; ++j) acc += gs[g * 40 + j] * inv * mlpW[j];
    out[g] = acc;
}

// ---------- host ----------
static inline int cld(long n, int b) { return (int)((n + b - 1) / b); }

extern "C" void kernel_launch(void* const* d_in, const int* in_sizes, int n_in,
                              void* d_out, int out_size, void* d_ws, size_t ws_size,
                              hipStream_t stream) {
    (void)in_sizes; (void)n_in; (void)out_size; (void)ws_size;
    const float* x     = (const float*)d_in[0];
    const int*   ei    = (const int*)d_in[1];
    const int*   batch = (const int*)d_in[2];
    const float* preW  = (const float*)d_in[3];
    const float* preb  = (const float*)d_in[4];
    const float* cprW  = (const float*)d_in[5];
    const float* cprb  = (const float*)d_in[6];
    const float* cpoW  = (const float*)d_in[7];
    const float* cpob  = (const float*)d_in[8];
    const float* clW   = (const float*)d_in[9];
    const float* clb   = (const float*)d_in[10];
    const float* gam   = (const float*)d_in[11];
    const float* bet   = (const float*)d_in[12];
    const float* mlpW  = (const float*)d_in[13];
    const float* mlpb  = (const float*)d_in[14];
    float* out = (float*)d_out;

    char* p = (char*)d_ws;
    auto take = [&](size_t bytes) -> char* {
        char* r = p;
        p += (bytes + 255) & ~(size_t)255;
        return r;
    };
    float*    op      = (float*)take((size_t)NN * H * 4);
    h16*      oph     = (h16*)  take((size_t)NN * KP * 2);
    float*    deg     = (float*)take((size_t)NN * 4);
    float*    amp     = (float*)take((size_t)NN * 4);
    float*    att     = (float*)take((size_t)NN * 4);
    float*    hnb     = (float*)take((size_t)NN * 4);
    float*    delta   = (float*)take(256);
    float*    hi      = (float*)take((size_t)NN * THW * 4);
    float*    hj      = (float*)take((size_t)NN * THW * 4);
    float*    sb      = (float*)take((size_t)NN * THW * 4);
    float*    s2b     = (float*)take((size_t)NN * THW * 4);
    unsigned* mnb     = (unsigned*)take((size_t)NN * THW * 4);
    unsigned* mxb     = (unsigned*)take((size_t)NN * THW * 4);
    h16*      wih     = (h16*)take((size_t)TW * 5 * 3 * 512 * 2);
    h16*      wjh     = (h16*)take((size_t)TW * 5 * 3 * 512 * 2);
    h16*      wph     = (h16*)take((size_t)TW * 33 * 512 * 2);
    h16*      wlh     = (h16*)take((size_t)5 * 3 * 512 * 2);
    h16*      outpost = (h16*)take((size_t)NN * KP * 2);
    float*    out2    = (float*)take((size_t)NN * H * 4);
    float*    bnstat  = (float*)take(2 * H * 4);
    float*    xc      = (float*)take((size_t)NN * 40 * 4);
    float*    ls      = (float*)take((size_t)NN * 40 * 4);
    float*    gs      = (float*)take((size_t)NG * 40 * 4);
    float*    gc      = (float*)take((size_t)NG * 4);

    const int B = 256;
    // prologue: op, degree stats
    k_fill_u32<<<cld(NN, B), B, 0, stream>>>((unsigned*)deg, 0u, NN);
    k_node_pre<<<cld((long)NN * KP, B), B, 0, stream>>>(x, preW, preb, op, oph);
    k_deg<<<cld(NE, B), B, 0, stream>>>(ei, deg);
    k_delta<<<1, 256, 0, stream>>>(deg, delta);
    k_scalars<<<cld(NN, B), B, 0, stream>>>(deg, delta, amp, att, hnb);

    for (int l = 0; l < 4; ++l) {
        k_cvt_pre<<<cld((long)TW * 5 * 3 * 512, B), B, 0, stream>>>(
            cprW + (size_t)l * TW * 2 * H * H, wih, wjh);
        k_cvt_post<<<cld((long)TW * 33 * 512, B), B, 0, stream>>>(
            cpoW + (size_t)l * TW * 13 * H * FO, wph);
        k_cvt_lin<<<cld((long)5 * 3 * 512, B), B, 0, stream>>>(clW + (size_t)l * H * H, wlh);

        k_fill_u32<<<cld((long)NN * THW, B), B, 0, stream>>>((unsigned*)sb, 0u, NN * THW);
        k_fill_u32<<<cld((long)NN * THW, B), B, 0, stream>>>((unsigned*)s2b, 0u, NN * THW);
        k_fill_u32<<<cld((long)NN * THW, B), B, 0, stream>>>(mnb, 0xFFFFFFFFu, NN * THW);
        k_fill_u32<<<cld((long)NN * THW, B), B, 0, stream>>>(mxb, 0u, NN * THW);
        k_fill_u32<<<cld(2 * H, B), B, 0, stream>>>((unsigned*)bnstat, 0u, 2 * H);
        k_fill_u32<<<cld((long)NN * KP / 2, B), B, 0, stream>>>((unsigned*)outpost, 0u,
                                                                NN * KP / 2);

        k_gemm_hihj<<<dim3(NN / 16, H / 16, TW * 2), 32, 0, stream>>>(oph, wih, wjh, hi, hj);
        k_edge_scatter<<<cld((long)NE * (THW / 4), B), B, 0, stream>>>(ei, hj, sb, s2b, mnb,
                                                                       mxb);
        k_post<<<dim3(NN / 16, TW), 32, 0, stream>>>(
            op, hi, cprb + (size_t)l * TW * H, deg, amp, att, hnb, sb, s2b, mnb, mxb, wph,
            cpob + (size_t)l * TW * FO, outpost);
        k_gemm_lin<<<dim3(NN / 16, H / 16), 32, 0, stream>>>(outpost, wlh,
                                                             clb + (size_t)l * H, out2);
        k_bnstat<<<cld((long)NN * H, B), B, 0, stream>>>(out2, bnstat);
        k_bnorm<<<cld((long)NN * KP, B), B, 0, stream>>>(out2, bnstat,
                                                         gam + (size_t)l * H,
                                                         bet + (size_t)l * H, op, oph);
    }

    // tail
    k_fill_u32<<<cld((long)NN * 40, B), B, 0, stream>>>((unsigned*)ls, 0u, NN * 40);
    k_fill_u32<<<cld(NG * 40, B), B, 0, stream>>>((unsigned*)gs, 0u, NG * 40);
    k_fill_u32<<<cld(NG, B), B, 0, stream>>>((unsigned*)gc, 0u, NG);
    k_xc<<<cld((long)NN * 40, B), B, 0, stream>>>(op, x, xc);
    k_logsum<<<cld((long)NE * 40, B), B, 0, stream>>>(ei, xc, ls);
    k_pool<<<cld(NN, B), B, 0, stream>>>(xc, ls, batch, gs, gc);
    k_final<<<1, 32, 0, stream>>>(gs, gc, mlpW, mlpb, out);
}
